// SDPAAttention_15461882265956
// MI455X (gfx1250) — compile-verified, matrix operand order checked by reading, not run on hardware
//
#include <hip/hip_runtime.h>
#include <hip/hip_bf16.h>

typedef __attribute__((ext_vector_type(16))) __bf16 v16bf;
typedef __attribute__((ext_vector_type(8)))  float  v8f;
typedef __attribute__((ext_vector_type(4)))  unsigned int v4u;
typedef __attribute__((ext_vector_type(8)))  int v8i;
typedef __attribute__((ext_vector_type(4)))  int v4i;

#define H_      16
#define HD_     64
#define DIM_    1024
#define NQ_     522        // new tokens / queries
#define KT_     18804      // total keys after cache surgery: 12 + 8*261 + 62*261 + 522
#define NEWOFF_ 18282      // where new k/v tokens land inside k_full/v_full
#define CTOK_   18282      // tokens gathered from the old cache
#define SCALE_  0.125f     // 64^-0.5

// ---------- bf16 helpers: use native cvt (v_cvt_*bf16*) instead of bit-twiddled RNE ----------
__device__ __forceinline__ unsigned short f2bf_hw(float f) {
  __bf16 b = (__bf16)f;
  unsigned short s; __builtin_memcpy(&s, &b, 2); return s;
}
__device__ __forceinline__ __bf16 bits2bf(unsigned short s) {
  __bf16 b; __builtin_memcpy(&b, &s, 2); return b;
}

// ---------- WMMA fragment loaders (16-bit A/B layout, ISA 7.12.2) ----------
// element e: K = (lane>=16 ? 8 : 0) + (e<8 ? e : 8 + e); caller passes row/col incl (lane&15)
__device__ __forceinline__ v16bf frag_from_f32(const float* src, int row, int ld, int kk) {
  int lane = threadIdx.x & 31;
  int kb = (lane >> 4) << 3;
  const float* p = src + (size_t)row * ld + kk + kb;
  v16bf f;
#pragma unroll
  for (int e = 0; e < 8; ++e) f[e] = (__bf16)p[e];
#pragma unroll
  for (int e = 0; e < 8; ++e) f[8 + e] = (__bf16)p[16 + e];
  return f;
}
__device__ __forceinline__ v16bf frag_from_u16(const unsigned short* src, int row, int ld, int kk) {
  int lane = threadIdx.x & 31;
  int kb = (lane >> 4) << 3;
  const unsigned short* p = src + (size_t)row * ld + kk + kb;
  v16bf f;
#pragma unroll
  for (int e = 0; e < 8; ++e) f[e] = bits2bf(p[e]);
#pragma unroll
  for (int e = 0; e < 8; ++e) f[8 + e] = bits2bf(p[16 + e]);
  return f;
}
// B-fragment where contraction (K) runs down rows of a row-major [K][N] tile (strided reads)
__device__ __forceinline__ v16bf frag_from_u16_kmajor(const unsigned short* src, int col, int ld) {
  int lane = threadIdx.x & 31;
  int kb = (lane >> 4) << 3;
  v16bf f;
#pragma unroll
  for (int e = 0; e < 8; ++e) f[e] = bits2bf(src[(kb + e) * ld + col]);
#pragma unroll
  for (int e = 0; e < 8; ++e) f[8 + e] = bits2bf(src[(kb + 16 + e) * ld + col]);
  return f;
}
__device__ __forceinline__ v8f wmma_bf16(v16bf a, v16bf b, v8f c) {
  return __builtin_amdgcn_wmma_f32_16x16x32_bf16(false, a, false, b, (short)0, c, false, false);
}

// ---------- Tensor Data Mover: 2D tile (rows x 64 bf16) global -> LDS ----------
// D# per CDNA5 ISA (8.3/8.4): group0 = {count=1, lds_addr, global_addr[56:0], type=2}
// group1 = {data_size=2B, tensor_dim0=64, tensor_dim1=rows_remaining (HW zero-fills OOB rows),
//           tile_dim0=64, tile_dim1=32, tensor_dim0_stride=64}
__device__ __forceinline__ void tdm_load_tile(unsigned lds_byte_addr,
                                              const unsigned short* gsrc,
                                              unsigned rows_remaining) {
  unsigned long long ga = (unsigned long long)(const void*)gsrc;
  v4u g0;
  g0[0] = 1u;                                              // count=1, no gather/iterate
  g0[1] = lds_byte_addr;                                   // LDS byte address
  g0[2] = (unsigned)ga;                                    // global_addr[31:0]
  g0[3] = (unsigned)((ga >> 32) & 0x01FFFFFFu) | (2u << 30); // global_addr[56:32] | type=2
  v8i g1;
  g1[0] = 0x00010000;                                      // workgroup_mask=0, data_size=1 (2B)
  g1[1] = (int)(64u << 16);                                // tensor_dim0[15:0]=64 in bits[63:48]
  g1[2] = (int)((rows_remaining & 0xFFFFu) << 16);         // tensor_dim1[15:0] in bits[95:80]
  g1[3] = (int)((rows_remaining >> 16) | (64u << 16));     // tensor_dim1[31:16] | tile_dim0=64
  g1[4] = 32;                                              // tile_dim1=32, tile_dim2=0
  g1[5] = 64;                                              // tensor_dim0_stride=64 (elements)
  g1[6] = 0;
  g1[7] = 0;
  v4i gz = {0, 0, 0, 0};
#if defined(__clang_major__) && (__clang_major__ >= 23)
  v8i gz8 = {0, 0, 0, 0, 0, 0, 0, 0};
  __builtin_amdgcn_tensor_load_to_lds(g0, g1, gz, gz, gz8, 0);
#else
  __builtin_amdgcn_tensor_load_to_lds(g0, g1, gz, gz, 0);
#endif
}

// ---------- kernel 1: gather evicted/kept cache tokens -> bf16 k_full/v_full ----------
__global__ __launch_bounds__(256) void gather_kv(const float* __restrict__ kc,
                                                 const float* __restrict__ vc,
                                                 unsigned short* __restrict__ kf,
                                                 unsigned short* __restrict__ vf) {
  int tid = blockIdx.x * blockDim.x + threadIdx.x;
  int d   = (tid & 7) * 8;
  int idx = tid >> 3;
  if (idx >= H_ * CTOK_) return;
  int h = idx / CTOK_;
  int t = idx - h * CTOK_;
  int frame, tok;
  if (t < 12)        { frame = 8 + t / 6;                       tok = t % 6; }
  else if (t < 2100) { int u = t - 12;   frame = u / 261;       tok = u - frame * 261; }
  else               { int u = t - 2100; frame = 10 + u / 261;  tok = u - (frame - 10) * 261; }
  size_t src = ((size_t)(h * 72 + frame) * 261 + tok) * HD_ + d;
  size_t dst = ((size_t)h * KT_ + t) * HD_ + d;
  union { unsigned short b[8]; uint4 u4; } pk;
#pragma unroll
  for (int j = 0; j < 8; ++j) pk.b[j] = f2bf_hw(kc[src + j]);
  *(uint4*)(kf + dst) = pk.u4;
#pragma unroll
  for (int j = 0; j < 8; ++j) pk.b[j] = f2bf_hw(vc[src + j]);
  *(uint4*)(vf + dst) = pk.u4;
}

// ---------- kernel 2: qkv = x @ W^T + b, routed to q_buf / k_full tail / v_full tail ----------
__global__ __launch_bounds__(128) void qkv_gemm(const float* __restrict__ x,
                                                const float* __restrict__ w,
                                                const float* __restrict__ bias,
                                                unsigned short* __restrict__ qb,
                                                unsigned short* __restrict__ kf,
                                                unsigned short* __restrict__ vf) {
  int wave = threadIdx.x >> 5, lane = threadIdx.x & 31;
  int ntile = blockIdx.x * 4 + wave;       // 0..191
  int mtile = blockIdx.y;                  // 0..32
  int nb = ntile * 16, mb = mtile * 16;
  int arow = mb + (lane & 15); if (arow >= NQ_) arow = NQ_ - 1;
  int brow = nb + (lane & 15);
  v8f c = {};
  for (int kk = 0; kk < DIM_; kk += 32) {
    v16bf a = frag_from_f32(x, arow, DIM_, kk);
    v16bf b = frag_from_f32(w, brow, DIM_, kk);   // B[k][n] = W[n][k]
    c = wmma_bf16(a, b, c);
  }
  int col = nb + (lane & 15);
  float bv = bias[col];
  int which = col >> 10, h = (col >> 6) & 15, d = col & 63;
  int hi8 = (lane >> 4) << 3;
#pragma unroll
  for (int s = 0; s < 8; ++s) {
    int m = mb + s + hi8;
    if (m >= NQ_) continue;
    unsigned short v = f2bf_hw(c[s] + bv);
    if (which == 0)      qb[((size_t)h * NQ_ + m) * HD_ + d] = v;
    else if (which == 1) kf[((size_t)h * KT_ + NEWOFF_ + m) * HD_ + d] = v;
    else                 vf[((size_t)h * KT_ + NEWOFF_ + m) * HD_ + d] = v;
  }
}

// ---------- kernel 3: flash attention; K/V tiles staged by the Tensor Data Mover ----------
__global__ __launch_bounds__(128) void attn_fwd(const unsigned short* __restrict__ qb,
                                                const unsigned short* __restrict__ kf,
                                                const unsigned short* __restrict__ vf,
                                                unsigned short* __restrict__ ob) {
  __shared__ unsigned short Kt[32 * 64];      // [key][d]
  __shared__ unsigned short Vr[32 * 64];      // [key][d] (raw; B-frags gather strided)
  __shared__ unsigned short Pt[4][16 * 32];   // per-wave P staging (C-layout -> A-layout)
  int h = blockIdx.y;
  int wave = threadIdx.x >> 5, lane = threadIdx.x & 31;
  int mb = blockIdx.x * 64 + wave * 16;
  const unsigned short* qh = qb + (size_t)h * NQ_ * HD_;
  const unsigned short* kh = kf + (size_t)h * KT_ * HD_;
  const unsigned short* vh = vf + (size_t)h * KT_ * HD_;

  unsigned ktLds = (unsigned)(unsigned long long)(void*)&Kt[0];
  unsigned vrLds = (unsigned)(unsigned long long)(void*)&Vr[0];

  int arow = mb + (lane & 15); if (arow >= NQ_) arow = NQ_ - 1;
  v16bf aq0 = frag_from_u16(qh, arow, HD_, 0);
  v16bf aq1 = frag_from_u16(qh, arow, HD_, 32);

  float mrow[8], lrow[8];
  v8f o0 = {}, o1 = {}, o2 = {}, o3 = {};
#pragma unroll
  for (int s = 0; s < 8; ++s) { mrow[s] = -3.0e38f; lrow[s] = 0.f; }
  int hi8 = (lane >> 4) << 3;

  const int NT = (KT_ + 31) / 32;             // 588 key tiles
  for (int t = 0; t < NT; ++t) {
    // TDM stages 32x64 bf16 tiles of K and V into LDS; HW zero-fills rows past KT_
    if (wave == 0) {
      unsigned rem = (unsigned)(KT_ - t * 32);
      tdm_load_tile(ktLds, kh + (size_t)t * 32 * HD_, rem);
      tdm_load_tile(vrLds, vh + (size_t)t * 32 * HD_, rem);
      __builtin_amdgcn_s_wait_tensorcnt(0);
    }
    __syncthreads();

    // S = Q K^T (two 16-key sub-tiles, contraction over d = 64 -> 2 chained WMMAs each)
    v16bf bk00 = frag_from_u16(Kt,      (lane & 15), 64, 0);
    v16bf bk01 = frag_from_u16(Kt,      (lane & 15), 64, 32);
    v16bf bk10 = frag_from_u16(Kt, 16 + (lane & 15), 64, 0);
    v16bf bk11 = frag_from_u16(Kt, 16 + (lane & 15), 64, 32);
    v8f c0 = {}, c1 = {};
    c0 = wmma_bf16(aq0, bk00, c0); c0 = wmma_bf16(aq1, bk01, c0);
    c1 = wmma_bf16(aq0, bk10, c1); c1 = wmma_bf16(aq1, bk11, c1);

    int k0g = t * 32 + (lane & 15), k1g = k0g + 16;
    unsigned short* pw = &Pt[wave][0];
#pragma unroll
    for (int s = 0; s < 8; ++s) {
      float s0 = (k0g < KT_) ? c0[s] * SCALE_ : -3.0e38f;
      float s1 = (k1g < KT_) ? c1[s] * SCALE_ : -3.0e38f;
      float tm = fmaxf(s0, s1);                // row reduce within 16-lane half
      tm = fmaxf(tm, __shfl_xor(tm, 1, 32));
      tm = fmaxf(tm, __shfl_xor(tm, 2, 32));
      tm = fmaxf(tm, __shfl_xor(tm, 4, 32));
      tm = fmaxf(tm, __shfl_xor(tm, 8, 32));
      float nm = fmaxf(mrow[s], tm);
      float corr = __expf(mrow[s] - nm);
      float p0 = __expf(s0 - nm);
      float p1 = __expf(s1 - nm);
      float rs = p0 + p1;
      rs += __shfl_xor(rs, 1, 32);
      rs += __shfl_xor(rs, 2, 32);
      rs += __shfl_xor(rs, 4, 32);
      rs += __shfl_xor(rs, 8, 32);
      lrow[s] = lrow[s] * corr + rs;
      mrow[s] = nm;
      o0[s] *= corr; o1[s] *= corr; o2[s] *= corr; o3[s] *= corr;
      pw[(s + hi8) * 32 +      (lane & 15)] = f2bf_hw(p0);
      pw[(s + hi8) * 32 + 16 + (lane & 15)] = f2bf_hw(p1);
    }
    // O += P V  (contraction over 32 keys; V B-frags gathered K-major from row-major tile)
    v16bf ap  = frag_from_u16(pw, (lane & 15), 32, 0);
    v16bf bv0 = frag_from_u16_kmajor(Vr,  0 + (lane & 15), 64);
    v16bf bv1 = frag_from_u16_kmajor(Vr, 16 + (lane & 15), 64);
    v16bf bv2 = frag_from_u16_kmajor(Vr, 32 + (lane & 15), 64);
    v16bf bv3 = frag_from_u16_kmajor(Vr, 48 + (lane & 15), 64);
    o0 = wmma_bf16(ap, bv0, o0);
    o1 = wmma_bf16(ap, bv1, o1);
    o2 = wmma_bf16(ap, bv2, o2);
    o3 = wmma_bf16(ap, bv3, o3);
    __syncthreads();
  }

  // epilogue: normalize and write o in (n, h*64+d) layout for proj GEMM
#pragma unroll
  for (int s = 0; s < 8; ++s) {
    int m = mb + s + hi8;
    if (m >= NQ_) continue;
    float inv = 1.0f / lrow[s];
    size_t base = (size_t)m * DIM_ + h * HD_ + (lane & 15);
    ob[base +  0] = f2bf_hw(o0[s] * inv);
    ob[base + 16] = f2bf_hw(o1[s] * inv);
    ob[base + 32] = f2bf_hw(o2[s] * inv);
    ob[base + 48] = f2bf_hw(o3[s] * inv);
  }
}

// ---------- kernel 4: out = o @ proj_w^T + proj_b (f32 output) ----------
__global__ __launch_bounds__(128) void proj_gemm(const unsigned short* __restrict__ ob,
                                                 const float* __restrict__ w,
                                                 const float* __restrict__ bias,
                                                 float* __restrict__ out) {
  int wave = threadIdx.x >> 5, lane = threadIdx.x & 31;
  int ntile = blockIdx.x * 4 + wave;       // 0..63
  int mtile = blockIdx.y;                  // 0..32
  int nb = ntile * 16, mb = mtile * 16;
  int arow = mb + (lane & 15); if (arow >= NQ_) arow = NQ_ - 1;
  int brow = nb + (lane & 15);
  v8f c = {};
  for (int kk = 0; kk < DIM_; kk += 32) {
    v16bf a = frag_from_u16(ob, arow, DIM_, kk);
    v16bf b = frag_from_f32(w, brow, DIM_, kk);
    c = wmma_bf16(a, b, c);
  }
  int col = nb + (lane & 15);
  float bv = bias[col];
  int hi8 = (lane >> 4) << 3;
#pragma unroll
  for (int s = 0; s < 8; ++s) {
    int m = mb + s + hi8;
    if (m < NQ_) out[(size_t)m * DIM_ + col] = c[s] + bv;
  }
}

extern "C" void kernel_launch(void* const* d_in, const int* in_sizes, int n_in,
                              void* d_out, int out_size, void* d_ws, size_t ws_size,
                              hipStream_t stream) {
  const float* x      = (const float*)d_in[0];
  const float* kcache = (const float*)d_in[1];
  const float* vcache = (const float*)d_in[2];
  const float* qkv_w  = (const float*)d_in[3];
  const float* qkv_b  = (const float*)d_in[4];
  const float* proj_w = (const float*)d_in[5];
  const float* proj_b = (const float*)d_in[6];
  float* out = (float*)d_out;

  // workspace layout (bf16 stored as u16): q_buf | k_full | v_full | o_buf  (~79 MB)
  const size_t QB = (size_t)H_ * NQ_ * HD_;
  const size_t KF = (size_t)H_ * KT_ * HD_;
  unsigned short* qb = (unsigned short*)d_ws;
  unsigned short* kf = qb + QB;
  unsigned short* vf = kf + KF;
  unsigned short* ob = vf + KF;

  int gthreads = H_ * CTOK_ * 8;
  gather_kv<<<dim3((gthreads + 255) / 256), 256, 0, stream>>>(kcache, vcache, kf, vf);
  qkv_gemm<<<dim3(48, 33), 128, 0, stream>>>(x, qkv_w, qkv_b, qb, kf, vf);
  attn_fwd<<<dim3(9, 16), 128, 0, stream>>>(qb, kf, vf, ob);
  proj_gemm<<<dim3(16, 33), 128, 0, stream>>>(ob, proj_w, proj_b, out);
}
